// SingleLayerSingleSpikeNet_19894288515265
// MI455X (gfx1250) — matrix operand order repaired and checked by reading, not compiled, not run on hardware
//
#include <hip/hip_runtime.h>
#include <hip/hip_bf16.h>
#include <stdint.h>

#define T_STEPS 100
#define BATCH   256
#define NIN     1024
#define NOUT    1024
#define KC      64      // K chunk staged in LDS per iteration
#define BT      32      // batch rows per workgroup
#define OT      64      // output cols per workgroup
#define FACTOR_F 0.9512294245007140f      // exp(-1e-7/2e-6)
#define OMF_F    0.0487705754992860f      // 1 - FACTOR
#define BIGV     99999.0f

typedef __bf16 bf16_t;
typedef __attribute__((ext_vector_type(16))) __bf16 v16bf;
typedef __attribute__((ext_vector_type(8)))  __bf16 v8bf;
typedef __attribute__((ext_vector_type(8)))  float  v8f;
typedef __attribute__((ext_vector_type(4)))  unsigned int u32x4;
typedef __attribute__((ext_vector_type(8)))  int          i32x8;
typedef __attribute__((ext_vector_type(4)))  int          i32x4;

union FragBF { v16bf v; v8bf h[2]; };

// ---- LDS layout (bytes) ----
#define XH_OFF 0                         // BT*KC bf16 = 4096 B
#define XL_OFF 4096
#define WH_OFF 8192                      // OT*KC bf16 = 8192 B
#define WL_OFF 16384
#define SMEM_BYTES 24576

// Issue a 2D TDM load: tile (tile_d1 rows x tile_d0 elems) of 2-byte data,
// row stride `stride0` elems, into LDS at byte offset lds_off (packed rows).
__device__ __forceinline__ void tdm_load_2d_bf16(unsigned lds_off, const bf16_t* gptr,
                                                 unsigned tile_d0, unsigned tile_d1,
                                                 unsigned tensor_d0, unsigned tensor_d1,
                                                 unsigned stride0)
{
    uint64_t ga = (uint64_t)(uintptr_t)gptr;
    u32x4 g0;
    g0[0] = 1u;                                   // count=1, user descriptor
    g0[1] = lds_off;                              // lds_addr (bytes)
    g0[2] = (unsigned)(ga & 0xFFFFFFFFu);         // global_addr[31:0]
    g0[3] = (unsigned)((ga >> 32) & 0x01FFFFFFu)  // global_addr[56:32]
          | (2u << 30);                           // type = 2 ("image")
    i32x8 g1;
    g1[0] = (int)(1u << 16);                      // wg_mask=0, data_size=1 (2B)
    g1[1] = (int)((tensor_d0 & 0xFFFFu) << 16);   // abar=0, tensor_dim0[15:0]
    g1[2] = (int)(((tensor_d0 >> 16) & 0xFFFFu) | ((tensor_d1 & 0xFFFFu) << 16));
    g1[3] = (int)(((tensor_d1 >> 16) & 0xFFFFu) | ((tile_d0 & 0xFFFFu) << 16));
    g1[4] = (int)(tile_d1 & 0xFFFFu);             // tile_dim1 ; tile_dim2 = 0
    g1[5] = (int)stride0;                         // tensor_dim0_stride[31:0]
    g1[6] = 0;                                    // stride hi + dim1_stride lo
    g1[7] = 0;
    i32x4 g2 = {0, 0, 0, 0};
    i32x4 g3 = {0, 0, 0, 0};
    i32x8 g4 = {0, 0, 0, 0, 0, 0, 0, 0};          // 6-arg toolchain: extra group
    __builtin_amdgcn_tensor_load_to_lds(g0, g1, g2, g3, g4, 0);
}

// Split W (f32) into bf16 hi/lo pair once.
__global__ void w_split_kernel(const float* __restrict__ W,
                               bf16_t* __restrict__ Wh, bf16_t* __restrict__ Wl)
{
    int i = blockIdx.x * blockDim.x + threadIdx.x;    // grid covers NOUT*NIN
    float w = W[i];
    bf16_t h = (bf16_t)w;
    Wh[i] = h;
    Wl[i] = (bf16_t)(w - (float)h);
}

__global__ __launch_bounds__(256)
void lif_fused_kernel(const float* __restrict__ x,
                      const bf16_t* __restrict__ Wh,
                      const bf16_t* __restrict__ Wl,
                      const float* __restrict__ ncvth,
                      const float* __restrict__ nciin,
                      const float* __restrict__ ndvth,
                      const float* __restrict__ ndrv,
                      const float* __restrict__ ndiin,
                      float* __restrict__ out)
{
    __shared__ __align__(128) unsigned char smem[SMEM_BYTES];
    bf16_t* xhS = (bf16_t*)(smem + XH_OFF);
    bf16_t* xlS = (bf16_t*)(smem + XL_OFF);
    bf16_t* whS = (bf16_t*)(smem + WH_OFF);
    bf16_t* wlS = (bf16_t*)(smem + WL_OFF);

    const int tid  = threadIdx.x;
    const int wave = tid >> 5;
    const int lane = tid & 31;
    const int hf   = lane >> 4;       // K-half selector per WMMA layout
    const int lr   = lane & 15;

    const int b0   = blockIdx.x * BT;               // gridDim.x = 8
    const int o0   = blockIdx.y * OT;               // gridDim.y = 16
    const int bsub = (wave & 1) * 16;
    const int osub = (wave >> 1) * 16;

    // ---- per-lane LIF constants (C/D layout: VGPR i -> row M = hf*8 + i, col N = lr)
    const int oc = o0 + osub + lr;
    const float chgRn = OMF_F * 100.0f * (1.0f + ndrv[oc]);
    const float dvth  = 1.0f + ndvth[oc];
    const float diin  = 1.0f + ndiin[oc];
    float vth[8], vr[8], gm[8], v[8];
    bool  spiked[8];
#pragma unroll
    for (int i = 0; i < 8; ++i) {
        const int bc = b0 + bsub + hf * 8 + i;
        const float cv = ncvth[(size_t)bc * NOUT + oc];
        float tv = dvth * (1.0f + cv);
        vth[i] = (tv <= 0.0f) ? BIGV : tv;
        float rv = 0.5f * (1.0f + cv);
        vr[i]  = (rv <= 0.0f) ? BIGV : rv;
        gm[i]  = chgRn * diin * (1.0f + nciin[(size_t)bc * NOUT + oc]);
        v[i] = 0.0f;
        spiked[i] = false;
    }

    // ---- x staging assignment: thread -> (row, 8 cols)
    const int xr = tid >> 3;          // 0..31
    const int xc = (tid & 7) * 8;     // 0..56

    for (int t = 0; t < T_STEPS; ++t) {
        v8f acc = {};                 // fresh cur accumulator each timestep
        for (int k = 0; k < NIN; k += KC) {
            // stage x[t, b0:b0+32, k:k+KC] split into bf16 hi/lo in LDS
            const float* xrow =
                x + ((size_t)t * BATCH + (b0 + xr)) * NIN + (k + xc);
            float4 f0 = *(const float4*)(xrow);
            float4 f1 = *(const float4*)(xrow + 4);
            if (k + KC < NIN) __builtin_prefetch(xrow + KC, 0, 0);
            float fv[8] = {f0.x, f0.y, f0.z, f0.w, f1.x, f1.y, f1.z, f1.w};
            v8bf hv, lv;
#pragma unroll
            for (int j = 0; j < 8; ++j) {
                bf16_t h = (bf16_t)fv[j];
                hv[j] = h;
                lv[j] = (bf16_t)(fv[j] - (float)h);
            }
            *(v8bf*)(xhS + xr * KC + xc) = hv;
            *(v8bf*)(xlS + xr * KC + xc) = lv;

            // TDM: DMA the (64 x KC) Wh/Wl tiles into LDS (wave 0 issues)
            if (wave == 0) {
                tdm_load_2d_bf16(WH_OFF, Wh + (size_t)o0 * NIN + k,
                                 KC, OT, NIN, 1u << 20, NIN);
                tdm_load_2d_bf16(WL_OFF, Wl + (size_t)o0 * NIN + k,
                                 KC, OT, NIN, 1u << 20, NIN);
                __builtin_amdgcn_s_wait_tensorcnt(0);
            }
            __syncthreads();

            // two K=32 WMMA steps per KC=64 chunk, 3 products each (bf16x3)
#pragma unroll
            for (int s = 0; s < 2; ++s) {
                const int ks = s * 32;
                FragBF ah, al, bh, bl;
                // A 16x32 bf16 layout: lanes 0-15 row=lr hold K {0..7,16..23},
                // lanes 16-31 same rows hold K {8..15,24..31}
                const bf16_t* ap = xhS + (bsub + lr) * KC + ks + hf * 8;
                ah.h[0] = *(const v8bf*)(ap);
                ah.h[1] = *(const v8bf*)(ap + 16);
                const bf16_t* alp = xlS + (bsub + lr) * KC + ks + hf * 8;
                al.h[0] = *(const v8bf*)(alp);
                al.h[1] = *(const v8bf*)(alp + 16);
                // B 32x16 bf16: lane col N=lr holds K {hf*8..+7, 16+hf*8..+7};
                // column N of B == row (o0+osub+N) of W -> contiguous in LDS
                const bf16_t* bp = whS + (osub + lr) * KC + ks + hf * 8;
                bh.h[0] = *(const v8bf*)(bp);
                bh.h[1] = *(const v8bf*)(bp + 16);
                const bf16_t* blp = wlS + (osub + lr) * KC + ks + hf * 8;
                bl.h[0] = *(const v8bf*)(blp);
                bl.h[1] = *(const v8bf*)(blp + 16);

                acc = __builtin_amdgcn_wmma_f32_16x16x32_bf16(
                          false, ah.v, false, bh.v, (short)0, acc, false, false);
                acc = __builtin_amdgcn_wmma_f32_16x16x32_bf16(
                          false, ah.v, false, bl.v, (short)0, acc, false, false);
                acc = __builtin_amdgcn_wmma_f32_16x16x32_bf16(
                          false, al.v, false, bh.v, (short)0, acc, false, false);
            }
            __syncthreads();
        }

        // ---- LIF dynamics + spike output for this timestep
#pragma unroll
        for (int i = 0; i < 8; ++i) {
            float cur = acc[i];
            float inc = spiked[i] ? 0.0f : gm[i] * cur;
            v[i] = FACTOR_F * v[i] + inc;
            float sp = (v[i] >= vth[i]) ? 1.0f : 0.0f;   // Heaviside fwd
            spiked[i] = spiked[i] || (sp != 0.0f);
            v[i] = (sp != 0.0f) ? vr[i] : v[i];          // detach reset
            out[((size_t)t * BATCH + (b0 + bsub + hf * 8 + i)) * NOUT + oc] = sp;
        }
    }
}

extern "C" void kernel_launch(void* const* d_in, const int* in_sizes, int n_in,
                              void* d_out, int out_size, void* d_ws, size_t ws_size,
                              hipStream_t stream)
{
    const float* x     = (const float*)d_in[0];   // [T,B,NIN]
    const float* W     = (const float*)d_in[1];   // [NOUT,NIN]
    const float* ncvth = (const float*)d_in[2];   // [B,NOUT]
    const float* nciin = (const float*)d_in[3];   // [B,NOUT]
    const float* ndvth = (const float*)d_in[4];   // [NOUT]
    const float* ndrv  = (const float*)d_in[5];   // [NOUT]
    const float* ndiin = (const float*)d_in[6];   // [NOUT]
    float* out = (float*)d_out;

    bf16_t* Wh = (bf16_t*)d_ws;                   // 2 MB
    bf16_t* Wl = Wh + (size_t)NOUT * NIN;         // 2 MB

    w_split_kernel<<<(NOUT * NIN) / 256, 256, 0, stream>>>(W, Wh, Wl);

    dim3 grid(BATCH / BT, NOUT / OT);             // 8 x 16 = 128 workgroups
    lif_fused_kernel<<<grid, 256, 0, stream>>>(x, Wh, Wl, ncvth, nciin,
                                               ndvth, ndrv, ndiin, out);
}
